// GPUPeakExtractor_57990648431143
// MI455X (gfx1250) — matrix operand order repaired and checked by reading, not compiled, not run on hardware
//
#include <hip/hip_runtime.h>

// ---------------------------------------------------------------------------
// GPUPeakExtractor for MI455X (gfx1250, wave32, WMMA)
//
// Memory-bound problem (134MB in, 96KB out). Strategy: tiled recompute with
// LDS staging; the frequency-direction Gaussian blur is executed as a banded
// 16x20 (K-padded to 32) matrix multiply on the matrix pipe via
// v_wmma_f32_16x16x32_f16. Ordered compaction uses count + scan + ballot-rank.
// ---------------------------------------------------------------------------

#define B_N     32
#define FREQ_N  256
#define TIME_N  4096
#define PADL    256

#define W_C     128              // output cols per chunk
#define NCHUNK  (TIME_N / W_C)   // 32
#define RBROWS  14               // output rows per row-block
#define NRB     19               // ceil(256/14)
#define QROWS   16               // q rows computed per block (f0-1 .. f0+14)
#define RAWROWS 22               // raw rows staged (f0-4 .. f0+17)
#define RAWW    146              // raw cols staged (t0-9 .. t0+136)
#define RAWSTR  148
#define QFW     144              // qfreq cols (t0-8 .. t0+135), 9 tiles of 16
#define P1K     32               // p1 rows (20 real + 12 zero pad for K=32)
#define P1STR   34
#define QW      130              // q cols (t0-1 .. t0+128)
#define QSTR    132

#define GW0 0.05448868f          // gaussian 5-tap, sigma=1, normalized
#define GW1 0.24420135f
#define GW2 0.40261995f

typedef __attribute__((ext_vector_type(16))) _Float16 v16h;
typedef __attribute__((ext_vector_type(8)))  float    v8f;

struct WS {
  unsigned mn1k[B_N], mx1k[B_N], mn2k[B_N], mx2k[B_N];
  int cellCount[B_N][FREQ_N][NCHUNK];
  int cellBase [B_N][FREQ_N][NCHUNK];
};

// order-preserving float <-> uint keys for atomic min/max
__device__ __forceinline__ unsigned fkey(float x){
  unsigned u = __float_as_uint(x);
  return (u & 0x80000000u) ? ~u : (u | 0x80000000u);
}
__device__ __forceinline__ float fdec(unsigned k){
  unsigned u = (k & 0x80000000u) ? (k & 0x7FFFFFFFu) : ~k;
  return __uint_as_float(u);
}
__device__ __forceinline__ int refl(int i, int n){
  if (i < 0)  return -i;
  if (i >= n) return 2 * n - 2 - i;
  return i;
}
__device__ __forceinline__ float bandw(int d){
  float w = 0.f;
  if (d == 2)              w = GW2;
  else if (d == 1 || d == 3) w = GW1;
  else if (d == 0 || d == 4) w = GW0;
  return w;
}

// ---------------------------------------------------------------------------
__global__ __launch_bounds__(256) void k_init(float* out, WS* ws){
  const int gid = blockIdx.x * 256 + threadIdx.x;
  if (gid < B_N * 3 * PADL) out[gid] = 0.f;
  if (gid < B_N){
    ws->mn1k[gid] = 0xFFFFFFFFu; ws->mx1k[gid] = 0u;
    ws->mn2k[gid] = 0xFFFFFFFFu; ws->mx2k[gid] = 0u;
  }
}

// per-batch min/max of raw input: one block per (row, batch)
__global__ __launch_bounds__(256) void k_minmax_raw(const float* __restrict__ x, WS* ws){
  __shared__ float smn[256], smx[256];
  const int b = blockIdx.y;
  const float* p = x + (size_t)b * FREQ_N * TIME_N + (size_t)blockIdx.x * TIME_N;
  float mn = INFINITY, mx = -INFINITY;
  for (int i = threadIdx.x; i < TIME_N; i += 256){
    const float v = p[i];
    mn = fminf(mn, v); mx = fmaxf(mx, v);
  }
  smn[threadIdx.x] = mn; smx[threadIdx.x] = mx;
  __syncthreads();
  for (int off = 128; off > 0; off >>= 1){
    if (threadIdx.x < off){
      smn[threadIdx.x] = fminf(smn[threadIdx.x], smn[threadIdx.x + off]);
      smx[threadIdx.x] = fmaxf(smx[threadIdx.x], smx[threadIdx.x + off]);
    }
    __syncthreads();
  }
  if (threadIdx.x == 0){
    atomicMin(&ws->mn1k[b], fkey(smn[0]));
    atomicMax(&ws->mx1k[b], fkey(smx[0]));
  }
}

// ---------------------------------------------------------------------------
// Core tile kernel: recompute peak1 -> blur (q) for a 14x128 output tile.
// MODE 0: reduce per-batch min/max of q
// MODE 1: count peak2 positives per (row, chunk)
// MODE 2: emit ordered compacted entries
template<int MODE>
__global__ __launch_bounds__(256) void k_tile(const float* __restrict__ x,
                                              float* __restrict__ out,
                                              WS* __restrict__ ws){
  __shared__ float    rawS[RAWROWS][RAWSTR];
  __shared__ _Float16 p1T [QFW][P1STR];       // transposed: [t][k-row], f16 for WMMA B
  __shared__ float    qfS [QROWS][QFW];       // freq-blurred (WMMA D tiles)
  __shared__ float    qS  [QROWS][QSTR];      // fully blurred
  __shared__ float    smn[256], smx[256];
  __shared__ int      cnt[RBROWS];
  __shared__ int      wcnt[8];

  const int chunk = blockIdx.x;
  const int rb    = blockIdx.y;
  const int b     = blockIdx.z;
  const int f0    = rb * RBROWS;
  const int t0    = chunk * W_C;
  const int tid   = threadIdx.x;
  const size_t base = (size_t)b * FREQ_N * TIME_N;

  const float mn1  = fdec(ws->mn1k[b]);
  const float mx1  = fdec(ws->mx1k[b]);
  const float inv1 = 1.0f / (mx1 - mn1);
  float mn2 = 0.f, inv2 = 0.f;
  if (MODE >= 1){
    mn2 = fdec(ws->mn2k[b]);
    inv2 = 1.0f / (fdec(ws->mx2k[b]) - mn2);
  }

  // ---- Stage 1: stage raw tile, OOB = -inf (matches SAME/-inf peak padding)
  for (int idx = tid; idx < RAWROWS * RAWW; idx += 256){
    const int j = idx / RAWW, i = idx - j * RAWW;
    const int r = f0 - 4 + j;
    const int c = t0 - 9 + i;
    float v = -INFINITY;
    if (r >= 0 && r < FREQ_N && c >= 0 && c < TIME_N)
      v = x[base + (size_t)r * TIME_N + c];
    rawS[j][i] = v;
  }
  __syncthreads();

  // ---- Stage 2: p1 = peak1 * (x-mn)/(mx-mn); reflect mapping folded into
  //      row/col indices; stored transposed f16 as the WMMA B operand.
  for (int idx = tid; idx < QFW * P1K; idx += 256){
    const int t = idx / P1K, k = idx - t * P1K;
    float p = 0.f;
    const int rr = f0 - 3 + k;                 // logical (padded) p1 row
    if (k < 20 && rr >= -2 && rr <= FREQ_N + 1){
      const int re = refl(rr, FREQ_N);
      const int te = refl(t0 - 8 + t, TIME_N);
      const int ri = re - (f0 - 4);
      const int ci = te - (t0 - 9);
      const float c  = rawS[ri][ci];
      const float mt = fmaxf(c, fmaxf(rawS[ri][ci-1], rawS[ri][ci+1]));
      const float mf = fmaxf(c, fmaxf(rawS[ri-1][ci], rawS[ri+1][ci]));
      if (c == mt && c == mf) p = (c - mn1) * inv1;
    }
    p1T[t][k] = (_Float16)p;                   // rows 20..31 stay exactly 0
  }
  __syncthreads();

  // ---- Stage 3: freq blur as banded matmul on the matrix pipe.
  //      qf[16 rows][16 cols] = A[16x32 band] x p1[32x16], K padded with zeros.
  {
    const int lane = tid & 31;
    const int wv   = tid >> 5;      // 8 waves
    const int m    = lane & 15;
    const int hi   = lane >> 4;
    v16h a;
    #pragma unroll
    for (int e = 0; e < 16; ++e){   // 16-bit A 16x32 layout: K per VGPR/half
      const int v  = e >> 1;
      const int kk = ((v < 4) ? (2 * v) : (16 + 2 * (v - 4))) + (e & 1) + 8 * hi;
      a[e] = (_Float16)bandw(kk - m);          // A[m][k] = w5[k-m], banded
    }
    for (int tile = wv; tile < QFW / 16; tile += 8){
      v16h bm;
      #pragma unroll
      for (int e = 0; e < 16; ++e)             // B 32x16: K = (e<8 ? e : e+8) + 8*hi, N = m
        bm[e] = p1T[tile * 16 + m][((e < 8) ? e : (e + 8)) + 8 * hi];
      v8f acc = {};
      acc = __builtin_amdgcn_wmma_f32_16x16x32_f16(false, a, false, bm,
                                                   (short)0, acc, false, false);
      #pragma unroll
      for (int j = 0; j < 8; ++j)              // D: row = j + 8*hi, col = m
        qfS[j + 8 * hi][tile * 16 + m] = acc[j];
    }
  }
  __syncthreads();

  // ---- Stage 4: time blur (reflect at global edges only)
  for (int idx = tid; idx < QROWS * QW; idx += 256){
    const int mr = idx / QW, i = idx - mr * QW;
    const int t  = t0 - 1 + i;
    float s;
    s  = GW0 * qfS[mr][refl(t - 2, TIME_N) - (t0 - 8)];
    s += GW1 * qfS[mr][refl(t - 1, TIME_N) - (t0 - 8)];
    s += GW2 * qfS[mr][refl(t,     TIME_N) - (t0 - 8)];
    s += GW1 * qfS[mr][refl(t + 1, TIME_N) - (t0 - 8)];
    s += GW0 * qfS[mr][refl(t + 2, TIME_N) - (t0 - 8)];
    qS[mr][i] = s;
  }
  __syncthreads();

  // ---- Stage 5 -------------------------------------------------------------
  if (MODE == 0){
    float mn = INFINITY, mx = -INFINITY;
    for (int idx = tid; idx < RBROWS * W_C; idx += 256){
      const int mr = idx / W_C, i = idx - mr * W_C;
      if (f0 + mr < FREQ_N){
        const float v = qS[mr + 1][i + 1];
        mn = fminf(mn, v); mx = fmaxf(mx, v);
      }
    }
    smn[tid] = mn; smx[tid] = mx;
    __syncthreads();
    for (int off = 128; off > 0; off >>= 1){
      if (tid < off){
        smn[tid] = fminf(smn[tid], smn[tid + off]);
        smx[tid] = fmaxf(smx[tid], smx[tid + off]);
      }
      __syncthreads();
    }
    if (tid == 0){
      atomicMin(&ws->mn2k[b], fkey(smn[0]));
      atomicMax(&ws->mx2k[b], fkey(smx[0]));
    }
  }

  if (MODE == 1){
    if (tid < RBROWS) cnt[tid] = 0;
    __syncthreads();
    for (int idx = tid; idx < RBROWS * W_C; idx += 256){
      const int mr = idx / W_C, i = idx - mr * W_C;
      const int r = f0 + mr, t = t0 + i;
      if (r < FREQ_N){
        const float q  = qS[mr + 1][i + 1];
        const float up = (r > 0)          ? qS[mr    ][i + 1] : -INFINITY;
        const float dn = (r + 1 < FREQ_N) ? qS[mr + 2][i + 1] : -INFINITY;
        const float lf = (t > 0)          ? qS[mr + 1][i    ] : -INFINITY;
        const float rt = (t + 1 < TIME_N) ? qS[mr + 1][i + 2] : -INFINITY;
        if (q >= up && q >= dn && q >= lf && q >= rt && q > mn2)
          atomicAdd(&cnt[mr], 1);
      }
    }
    __syncthreads();
    if (tid < RBROWS && f0 + tid < FREQ_N)
      ws->cellCount[b][f0 + tid][chunk] = cnt[tid];
  }

  if (MODE == 2){
    const int lane = tid & 31;
    const int wv   = tid >> 5;
    for (int mr = 0; mr < RBROWS; ++mr){
      const int r = f0 + mr;
      bool pred = false; float val = 0.f; int t = 0;
      if (r < FREQ_N && tid < W_C){
        t = t0 + tid;
        const int i = tid + 1;
        const float q  = qS[mr + 1][i];
        const float up = (r > 0)          ? qS[mr    ][i    ] : -INFINITY;
        const float dn = (r + 1 < FREQ_N) ? qS[mr + 2][i    ] : -INFINITY;
        const float lf = (t > 0)          ? qS[mr + 1][i - 1] : -INFINITY;
        const float rt = (t + 1 < TIME_N) ? qS[mr + 1][i + 1] : -INFINITY;
        pred = (q >= up && q >= dn && q >= lf && q >= rt && q > mn2);
        val  = (q - mn2) * inv2;
      }
      const unsigned bal = (unsigned)__ballot(pred);   // wave32: low 32 bits
      if (lane == 0) wcnt[wv] = __popc(bal);
      __syncthreads();
      if (pred){
        int rank = __popc(bal & ((1u << lane) - 1u));
        for (int w = 0; w < wv; ++w) rank += wcnt[w];  // waves 4..7 contribute 0
        const int pos = ws->cellBase[b][r][chunk] + rank;
        if (pos < PADL){
          float* ob = out + (size_t)b * 3 * PADL;
          ob[0 * PADL + pos] = (float)r * (1.0f / FREQ_N);
          ob[1 * PADL + pos] = (float)t * (1.0f / TIME_N);
          ob[2 * PADL + pos] = val;
        }
      }
      __syncthreads();
    }
  }
}

// per-batch exclusive scan of the 8192 (row-major) cell counts
__global__ __launch_bounds__(256) void k_scan(WS* ws){
  __shared__ int tsum[256];
  const int b = blockIdx.x;
  const int N = FREQ_N * NCHUNK;     // 8192
  const int PER = N / 256;           // 32
  int* cc = &ws->cellCount[b][0][0];
  int* cb = &ws->cellBase [b][0][0];
  const int tid = threadIdx.x;
  const int base = tid * PER;
  int s = 0;
  for (int i = 0; i < PER; ++i) s += cc[base + i];
  tsum[tid] = s;
  __syncthreads();
  for (int off = 1; off < 256; off <<= 1){
    const int v   = tsum[tid];
    const int add = (tid >= off) ? tsum[tid - off] : 0;
    __syncthreads();
    tsum[tid] = v + add;
    __syncthreads();
  }
  int run = (tid == 0) ? 0 : tsum[tid - 1];
  for (int i = 0; i < PER; ++i){ cb[base + i] = run; run += cc[base + i]; }
}

// ---------------------------------------------------------------------------
extern "C" void kernel_launch(void* const* d_in, const int* in_sizes, int n_in,
                              void* d_out, int out_size, void* d_ws, size_t ws_size,
                              hipStream_t stream){
  (void)in_sizes; (void)n_in; (void)out_size; (void)ws_size;
  const float* x = (const float*)d_in[0];
  float* out = (float*)d_out;
  WS* ws = (WS*)d_ws;                       // needs ~2.1 MB of workspace

  const dim3 blk(256);
  k_init<<<dim3((B_N * 3 * PADL + 255) / 256), blk, 0, stream>>>(out, ws);
  k_minmax_raw<<<dim3(FREQ_N, B_N), blk, 0, stream>>>(x, ws);

  const dim3 grid(NCHUNK, NRB, B_N);
  k_tile<0><<<grid, blk, 0, stream>>>(x, out, ws);  // min/max of blurred
  k_tile<1><<<grid, blk, 0, stream>>>(x, out, ws);  // per-cell counts
  k_scan<<<dim3(B_N), blk, 0, stream>>>(ws);        // exclusive scan
  k_tile<2><<<grid, blk, 0, stream>>>(x, out, ws);  // ordered emit
}
